// MultiHeadedAttention_89043261981427
// MI455X (gfx1250) — compile-verified, matrix operand order checked by reading, not run on hardware
//
#include <hip/hip_runtime.h>
#include <hip/hip_bf16.h>

// ---------------------------------------------------------------------------
// MultiHeadedAttention for MI455X (gfx1250).
//  - All matmuls via v_wmma_f32_16x16x32_bf16 (wave32), f32 accumulate.
//  - All LDS tiles stored so A and B fragments load as 2x ds_load_b128.
//  - Attention K/V tiles streamed by the Tensor Data Mover, double-buffered
//    so the DMA overlaps the WMMA + softmax work of the current tile.
//  - pool_score A-tiles (already bf16) also staged via TDM.
// ---------------------------------------------------------------------------

#define B_  8
#define S_  1024
#define D_  1024
#define H_  16
#define DK_ 64

typedef __attribute__((ext_vector_type(16))) __bf16       v16bf;
typedef __attribute__((ext_vector_type(8)))  float        v8f;
typedef __attribute__((ext_vector_type(4)))  unsigned int u32x4;
typedef __attribute__((ext_vector_type(4)))  int          i32x4;
typedef __attribute__((ext_vector_type(8)))  int          i32x8;

static __device__ inline v8f wmma_bf16(v16bf a, v16bf b, v8f c) {
  return __builtin_amdgcn_wmma_f32_16x16x32_bf16(false, a, false, b,
                                                 (short)0, c, false, false);
}

// A-operand (16x32 MxK) from LDS [m][ld]: lane row = lane%16,
// elem e -> k = (e/8)*16 + (lane/16)*8 + (e%8)  => two contiguous 8-elem runs.
static __device__ inline v16bf load_frag_a(const __bf16* smem, int ld,
                                           int row_base, int k_base, int lane) {
  v16bf a;
  const int m = row_base + (lane & 15);
  const int half = lane >> 4;
#pragma unroll
  for (int e = 0; e < 16; ++e) {
    const int k = k_base + ((e >> 3) << 4) + (half << 3) + (e & 7);
    a[e] = smem[m * ld + k];
  }
  return a;
}

// B-operand (32x16 KxN) from LDS stored n-major [n][ld] -> same contiguous runs.
static __device__ inline v16bf load_frag_bt(const __bf16* smem, int ld,
                                            int k_base, int col_base, int lane) {
  v16bf b;
  const int n = col_base + (lane & 15);
  const int half = lane >> 4;
#pragma unroll
  for (int e = 0; e < 16; ++e) {
    const int k = k_base + ((e >> 3) << 4) + (half << 3) + (e & 7);
    b[e] = smem[n * ld + k];
  }
  return b;
}

// ---------------------------------------------------------------------------
// Tensor Data Mover: 2D tile (elem=2B) global -> LDS with LDS row padding.
//  pad_int_code: pad after 2<<code dwords; pad_amt_code: pad (code+1) dwords.
// ---------------------------------------------------------------------------
static __device__ inline void tdm_load_2d_bf16(unsigned lds_addr, const void* gptr,
                                               unsigned tile_w, unsigned tile_h,
                                               unsigned row_stride_elems,
                                               unsigned tensor_rows,
                                               unsigned pad_int_code,
                                               unsigned pad_amt_code) {
  const unsigned long long ga = (unsigned long long)(uintptr_t)gptr;
  u32x4 g0;
  g0[0] = 1u;                                   // count = 1 valid descriptor
  g0[1] = lds_addr;                             // LDS byte address
  g0[2] = (unsigned)ga;                         // global_addr[31:0]
  g0[3] = (unsigned)((ga >> 32) & 0x1FFFFFFu)   // global_addr[56:32]
        | (2u << 30);                           // type = 2 ("image")
  i32x8 g1;
  g1[0] = (int)((1u << 16)                      // data_size = 1 -> 2 bytes
              | (1u << 20)                      // pad_enable
              | (pad_int_code << 22)
              | (pad_amt_code << 25));
  g1[1] = (int)(row_stride_elems << 16);        // tensor_dim0 low16
  g1[2] = (int)((tensor_rows & 0xffffu) << 16); // dim0 hi=0 | tensor_dim1 lo16
  g1[3] = (int)(tile_w << 16);                  // dim1 hi=0 | tile_dim0
  g1[4] = (int)tile_h;                          // tile_dim1 | tile_dim2=0
  g1[5] = (int)row_stride_elems;                // tensor_dim0_stride[31:0]
  g1[6] = 0;
  g1[7] = 0;
  const i32x4 z4 = {0, 0, 0, 0};
#if defined(__clang_major__) && (__clang_major__ >= 23)
  const i32x8 z8 = {0, 0, 0, 0, 0, 0, 0, 0};
  __builtin_amdgcn_tensor_load_to_lds(g0, g1, z4, z4, z8, 0);
#else
  __builtin_amdgcn_tensor_load_to_lds(g0, g1, z4, z4, 0);
#endif
}

static __device__ inline unsigned lds_byte_addr(const void* p) {
  // Generic LDS-aperture address: low 32 bits are the LDS byte offset.
  return (unsigned)(uintptr_t)p;
}

// ---------------------------------------------------------------------------
// Kernel 1: fused QKV projection  y = x @ W^T + b.
//  q,k -> [B,H,S,DK] bf16 ; v -> [B,H,DK,S] bf16 (transposed for P@V B-operand)
//  grid = (8192/128, 1024/64, 3), block = 256 (8 waves, 4x2)
// ---------------------------------------------------------------------------
#define LDA 40  // 32 + 8 pad (bf16 elems); keeps 16B alignment for b128 frags

__global__ __launch_bounds__(256) void qkv_proj_kernel(
    const float* __restrict__ q_in, const float* __restrict__ k_in,
    const float* __restrict__ v_in,
    const float* __restrict__ Wq, const float* __restrict__ bq,
    const float* __restrict__ Wk, const float* __restrict__ bk,
    const float* __restrict__ Wv, const float* __restrict__ bv,
    __bf16* __restrict__ qb, __bf16* __restrict__ kb, __bf16* __restrict__ vt) {
  const float* X; const float* W; const float* bias; __bf16* Y;
  const bool isV = (blockIdx.z == 2);
  if (blockIdx.z == 0)      { X = q_in; W = Wq; bias = bq; Y = qb; }
  else if (blockIdx.z == 1) { X = k_in; W = Wk; bias = bk; Y = kb; }
  else                      { X = v_in; W = Wv; bias = bv; Y = vt; }

  const int m0 = blockIdx.x * 128;
  const int n0 = blockIdx.y * 64;
  const int tid = threadIdx.x;
  const int wave = tid >> 5, lane = tid & 31;
  const int wm = wave >> 1, wn = wave & 1;
  const int half = lane >> 4, ln = lane & 15;

  __shared__ __align__(16) __bf16 As[128 * LDA];  // [m][k]
  __shared__ __align__(16) __bf16 Bs[64 * LDA];   // [n][k]

  v8f acc[2][2];
#pragma unroll
  for (int i = 0; i < 2; ++i)
#pragma unroll
    for (int j = 0; j < 2; ++j)
      acc[i][j] = (v8f){0.f,0.f,0.f,0.f,0.f,0.f,0.f,0.f};

  for (int kk = 0; kk < D_; kk += 32) {
    // A tile: 128x32 fp32 -> bf16, [m][k]
    for (int i = tid; i < 1024; i += 256) {
      const int row = i >> 3, c4 = (i & 7) << 2;
      const float4 f = *(const float4*)(X + (size_t)(m0 + row) * D_ + kk + c4);
      __bf16* d = As + row * LDA + c4;
      d[0] = (__bf16)f.x; d[1] = (__bf16)f.y; d[2] = (__bf16)f.z; d[3] = (__bf16)f.w;
    }
    // B tile: 64x32, n-major contiguous copy of W rows
    for (int i = tid; i < 512; i += 256) {
      const int n = i >> 3, kq = (i & 7) << 2;
      const float4 f = *(const float4*)(W + (size_t)(n0 + n) * D_ + kk + kq);
      __bf16* d = Bs + n * LDA + kq;
      d[0] = (__bf16)f.x; d[1] = (__bf16)f.y; d[2] = (__bf16)f.z; d[3] = (__bf16)f.w;
    }
    if (kk + 32 < D_) {
      __builtin_prefetch(X + (size_t)(m0 + (tid >> 1)) * D_ + kk + 32, 0, 1);
      __builtin_prefetch(W + (size_t)(n0 + (tid & 63)) * D_ + kk + 32, 0, 1);
    }
    __syncthreads();

    v16bf bf[2];
#pragma unroll
    for (int j = 0; j < 2; ++j)
      bf[j] = load_frag_bt(Bs, LDA, 0, wn * 32 + j * 16, lane);
#pragma unroll
    for (int i = 0; i < 2; ++i) {
      const v16bf af = load_frag_a(As, LDA, wm * 32 + i * 16, 0, lane);
#pragma unroll
      for (int j = 0; j < 2; ++j)
        acc[i][j] = wmma_bf16(af, bf[j], acc[i][j]);
    }
    __syncthreads();
  }

#pragma unroll
  for (int i = 0; i < 2; ++i)
#pragma unroll
    for (int j = 0; j < 2; ++j)
#pragma unroll
      for (int r = 0; r < 8; ++r) {
        const int row = m0 + wm * 32 + i * 16 + r + half * 8;  // b*S + s
        const int col = n0 + wn * 32 + j * 16 + ln;            // h*64 + dk
        const float v = acc[i][j][r] + bias[col];
        const int bidx = row >> 10, s = row & (S_ - 1);
        const int h = col >> 6, dk = col & (DK_ - 1);
        const size_t off = isV
            ? (((size_t)(bidx * H_ + h)) * DK_ + dk) * S_ + s    // [B,H,DK,S]
            : (((size_t)(bidx * H_ + h)) * S_ + s) * DK_ + dk;   // [B,H,S,DK]
        Y[off] = (__bf16)v;
      }
}

// ---------------------------------------------------------------------------
// Kernel 2: flash attention per (b, h, 64-query tile), TDM double-buffered
//  grid = (S/64, H, B), block = 128 (4 waves x 16 query rows)
// ---------------------------------------------------------------------------
#define LDT 72  // 64 + 8 pad (bf16 elems), matches TDM pad (4 dw per 32 dw)

__global__ __launch_bounds__(128) void attn_kernel(
    const __bf16* __restrict__ qb, const __bf16* __restrict__ kb,
    const __bf16* __restrict__ vt,
    float* __restrict__ x, __bf16* __restrict__ xb) {
  const int qt = blockIdx.x;
  const int h  = blockIdx.y;
  const int b  = blockIdx.z;
  const int tid = threadIdx.x;
  const int wave = tid >> 5, lane = tid & 31;
  const int half = lane >> 4, ln = lane & 15;

  __shared__ __align__(16) __bf16 Qs[64 * LDT];      // [q][dk]
  __shared__ __align__(16) __bf16 Ks[2][64 * LDT];   // [key][dk]  (B for scores)
  __shared__ __align__(16) __bf16 Vts[2][64 * LDT];  // [dk][key]  (B for P@V)
  __shared__ __align__(16) __bf16 Ps[64 * LDT];      // [q][key]   (A for P@V)

  const size_t head_off = ((size_t)(b * H_ + h)) * S_ * DK_;
  const __bf16* qg = qb + head_off + (size_t)qt * 64 * DK_;

  // Stage Q once (vectorized copy into padded layout)
  for (int i = tid; i < 512; i += 128) {          // 64*64/8 uint4
    const int q = i >> 3, c8 = (i & 7) << 3;
    *(uint4*)(Qs + q * LDT + c8) = *(const uint4*)(qg + q * DK_ + c8);
  }
  // Prologue: TDM tile 0 into buffer 0
  if (wave == 0) {
    tdm_load_2d_bf16(lds_byte_addr(Ks[0]),  kb + head_off, 64, 64, DK_, 64, 4, 3);
    tdm_load_2d_bf16(lds_byte_addr(Vts[0]), vt + head_off, 64, 64, S_,  64, 4, 3);
    __builtin_amdgcn_s_wait_tensorcnt(0);
  }
  __syncthreads();  // covers Q staging + tile 0

  v16bf qf[2];
  qf[0] = load_frag_a(Qs, LDT, wave * 16, 0, lane);
  qf[1] = load_frag_a(Qs, LDT, wave * 16, 32, lane);

  v8f o[4];
#pragma unroll
  for (int j = 0; j < 4; ++j) o[j] = (v8f){0.f,0.f,0.f,0.f,0.f,0.f,0.f,0.f};
  float mrow[8], lrow[8];
#pragma unroll
  for (int r = 0; r < 8; ++r) { mrow[r] = -1e30f; lrow[r] = 0.f; }

  const float scale = 0.125f;  // 1/sqrt(DK)
  const int NT = S_ / 64;

  for (int kt = 0; kt < NT; ++kt) {
    const int cur = kt & 1;
    // Prefetch next tile into the other buffer (its consumers retired at the
    // barrier that ended iteration kt-1) -- overlaps with this tile's WMMAs.
    if (wave == 0 && kt + 1 < NT) {
      tdm_load_2d_bf16(lds_byte_addr(Ks[1 - cur]),
                       kb + head_off + (size_t)(kt + 1) * 64 * DK_,
                       64, 64, DK_, 64, 4, 3);
      tdm_load_2d_bf16(lds_byte_addr(Vts[1 - cur]),
                       vt + head_off + (size_t)(kt + 1) * 64,
                       64, 64, S_, 64, 4, 3);
    }
    const __bf16* Kc  = Ks[cur];
    const __bf16* Vtc = Vts[cur];

    // scores = Q @ K^T  (B-operand n-major from Kc[key][dk])
    v8f sc[4];
#pragma unroll
    for (int jn = 0; jn < 4; ++jn) {
      v8f c = (v8f){0.f,0.f,0.f,0.f,0.f,0.f,0.f,0.f};
#pragma unroll
      for (int kc = 0; kc < 2; ++kc) {
        const v16bf bf = load_frag_bt(Kc, LDT, kc * 32, jn * 16, lane);
        c = wmma_bf16(qf[kc], bf, c);
      }
#pragma unroll
      for (int r = 0; r < 8; ++r) c[r] *= scale;
      sc[jn] = c;
    }

    // Online softmax (C layout: reg r -> row r+8*half, lanes -> 16 cols)
    float rsc[8];
#pragma unroll
    for (int r = 0; r < 8; ++r) {
      float v = fmaxf(fmaxf(sc[0][r], sc[1][r]), fmaxf(sc[2][r], sc[3][r]));
      for (int m = 1; m <= 8; m <<= 1) v = fmaxf(v, __shfl_xor(v, m, 32));
      const float nm = fmaxf(mrow[r], v);
      rsc[r] = __expf(mrow[r] - nm);
      mrow[r] = nm;
      lrow[r] *= rsc[r];
    }
#pragma unroll
    for (int r = 0; r < 8; ++r) {
      float s = 0.f;
#pragma unroll
      for (int jn = 0; jn < 4; ++jn) {
        const float p = __expf(sc[jn][r] - mrow[r]);
        sc[jn][r] = p;
        s += p;
      }
      for (int m = 1; m <= 8; m <<= 1) s += __shfl_xor(s, m, 32);
      lrow[r] += s;
    }

    // P tile to LDS (wave-private rows; per-wave DS ordering handles RAW)
#pragma unroll
    for (int jn = 0; jn < 4; ++jn)
#pragma unroll
      for (int r = 0; r < 8; ++r)
        Ps[(wave * 16 + r + half * 8) * LDT + jn * 16 + ln] = (__bf16)sc[jn][r];

#pragma unroll
    for (int jd = 0; jd < 4; ++jd)
#pragma unroll
      for (int r = 0; r < 8; ++r) o[jd][r] *= rsc[r];

    // O += P @ V  (A from Ps, B n-major from Vtc[dk][key])
#pragma unroll
    for (int kc = 0; kc < 2; ++kc) {
      const v16bf af = load_frag_a(Ps, LDT, wave * 16, kc * 32, lane);
#pragma unroll
      for (int jd = 0; jd < 4; ++jd) {
        const v16bf bf = load_frag_bt(Vtc, LDT, kc * 32, jd * 16, lane);
        o[jd] = wmma_bf16(af, bf, o[jd]);
      }
    }

    __syncthreads();  // all waves done with buffer `cur`
    if (wave == 0 && kt + 1 < NT) __builtin_amdgcn_s_wait_tensorcnt(0);
    __syncthreads();  // next tile's TDM data visible to all waves
  }

  // Epilogue: normalize and write fp32 x + bf16 xb in [B,S,D]
#pragma unroll
  for (int jd = 0; jd < 4; ++jd)
#pragma unroll
    for (int r = 0; r < 8; ++r) {
      const int s = qt * 64 + wave * 16 + r + half * 8;
      const int d = h * DK_ + jd * 16 + ln;
      const float v = o[jd][r] / lrow[r];
      const size_t off = ((size_t)(b * S_ + s)) * D_ + d;
      x[off] = v;
      xb[off] = (__bf16)v;
    }
}

// ---------------------------------------------------------------------------
// Kernel 3: a[b,s] = sum_n tanh(x @ Wa^T + ba)[.,n] * qvec[n]
//  grid = 8192/128, block = 256 (8 waves, 4x2). A-tile staged via TDM.
// ---------------------------------------------------------------------------
__global__ __launch_bounds__(256) void pool_score_kernel(
    const __bf16* __restrict__ xb, const float* __restrict__ Wa,
    const float* __restrict__ ba, const float* __restrict__ qvec,
    float* __restrict__ a_out) {
  const int m0 = blockIdx.x * 128;
  const int tid = threadIdx.x;
  const int wave = tid >> 5, lane = tid & 31;
  const int wm = wave >> 1, wn = wave & 1;
  const int half = lane >> 4, ln = lane & 15;

  __shared__ __align__(16) __bf16 As[128 * LDA];
  __shared__ __align__(16) __bf16 Bs[64 * LDA];
  __shared__ float a_acc[128];

  for (int i = tid; i < 128; i += 256) a_acc[i] = 0.f;

  for (int nc = 0; nc < D_ / 64; ++nc) {
    const int n0 = nc * 64;
    v8f acc[2][2];
#pragma unroll
    for (int i = 0; i < 2; ++i)
#pragma unroll
      for (int j = 0; j < 2; ++j)
        acc[i][j] = (v8f){0.f,0.f,0.f,0.f,0.f,0.f,0.f,0.f};

    for (int kk = 0; kk < D_; kk += 32) {
      __syncthreads();
      // A tile (bf16, 128x32, row stride D): TDM with pad 4 dw per 16 dw -> LDA=40
      if (wave == 0)
        tdm_load_2d_bf16(lds_byte_addr(As), xb + (size_t)m0 * D_ + kk,
                         32, 128, D_, 128, 3, 3);
      // B tile: Wa rows, n-major, fp32 -> bf16 (overlaps the TDM transfer)
      for (int i = tid; i < 512; i += 256) {
        const int n = i >> 3, kq = (i & 7) << 2;
        const float4 f = *(const float4*)(Wa + (size_t)(n0 + n) * D_ + kk + kq);
        __bf16* d = Bs + n * LDA + kq;
        d[0] = (__bf16)f.x; d[1] = (__bf16)f.y; d[2] = (__bf16)f.z; d[3] = (__bf16)f.w;
      }
      if (kk + 32 < D_)
        __builtin_prefetch(xb + (size_t)(m0 + (tid >> 1)) * D_ + kk + 32, 0, 1);
      if (wave == 0) __builtin_amdgcn_s_wait_tensorcnt(0);
      __syncthreads();

      v16bf bf[2];
#pragma unroll
      for (int j = 0; j < 2; ++j)
        bf[j] = load_frag_bt(Bs, LDA, 0, wn * 32 + j * 16, lane);
#pragma unroll
      for (int i = 0; i < 2; ++i) {
        const v16bf af = load_frag_a(As, LDA, wm * 32 + i * 16, 0, lane);
#pragma unroll
        for (int j = 0; j < 2; ++j)
          acc[i][j] = wmma_bf16(af, bf[j], acc[i][j]);
      }
    }

    // Fused epilogue: tanh(.+ba)*qvec, accumulate per-row in LDS (ds_add_f32)
#pragma unroll
    for (int i = 0; i < 2; ++i)
#pragma unroll
      for (int r = 0; r < 8; ++r) {
        float t = 0.f;
#pragma unroll
        for (int j = 0; j < 2; ++j) {
          const int col = n0 + wn * 32 + j * 16 + ln;
          t += tanhf(acc[i][j][r] + ba[col]) * qvec[col];
        }
        atomicAdd(&a_acc[wm * 32 + i * 16 + r + half * 8], t);
      }
  }
  __syncthreads();
  for (int i = tid; i < 128; i += 256) a_out[m0 + i] = a_acc[i];
}

// ---------------------------------------------------------------------------
// Kernel 4: alpha = softmax(a) over S, per batch.  grid = B, block = 256
// ---------------------------------------------------------------------------
__global__ __launch_bounds__(256) void alpha_softmax_kernel(
    const float* __restrict__ a, float* __restrict__ alpha) {
  const int b = blockIdx.x;
  const int tid = threadIdx.x;
  __shared__ float red[256];

  float m = -1e30f;
  for (int s = tid; s < S_; s += 256) m = fmaxf(m, a[b * S_ + s]);
  red[tid] = m;
  __syncthreads();
  for (int st = 128; st > 0; st >>= 1) {
    if (tid < st) red[tid] = fmaxf(red[tid], red[tid + st]);
    __syncthreads();
  }
  m = red[0];
  __syncthreads();

  float sum = 0.f;
  for (int s = tid; s < S_; s += 256) sum += __expf(a[b * S_ + s] - m);
  red[tid] = sum;
  __syncthreads();
  for (int st = 128; st > 0; st >>= 1) {
    if (tid < st) red[tid] += red[tid + st];
    __syncthreads();
  }
  const float inv = 1.0f / red[0];

  for (int s = tid; s < S_; s += 256)
    alpha[b * S_ + s] = __expf(a[b * S_ + s] - m) * inv;
}

// ---------------------------------------------------------------------------
// Kernel 5: out[b,d] = sum_s alpha[b,s] * x[b,s,d].  grid = (D/256, B)
// ---------------------------------------------------------------------------
__global__ __launch_bounds__(256) void pool_reduce_kernel(
    const float* __restrict__ alpha, const float* __restrict__ x,
    float* __restrict__ out) {
  const int b = blockIdx.y;
  const int d = blockIdx.x * 256 + threadIdx.x;
  float acc = 0.f;
  for (int s = 0; s < S_; ++s)
    acc += alpha[b * S_ + s] * x[((size_t)(b * S_ + s)) * D_ + d];
  out[b * D_ + d] = acc;
}

// ---------------------------------------------------------------------------
extern "C" void kernel_launch(void* const* d_in, const int* in_sizes, int n_in,
                              void* d_out, int out_size, void* d_ws, size_t ws_size,
                              hipStream_t stream) {
  (void)in_sizes; (void)n_in; (void)out_size; (void)ws_size;
  const float* query = (const float*)d_in[0];
  const float* key_  = (const float*)d_in[1];
  const float* value = (const float*)d_in[2];
  const float* Wq = (const float*)d_in[3];
  const float* bq = (const float*)d_in[4];
  const float* Wk = (const float*)d_in[5];
  const float* bk = (const float*)d_in[6];
  const float* Wv = (const float*)d_in[7];
  const float* bv = (const float*)d_in[8];
  const float* Wa = (const float*)d_in[9];
  const float* ba = (const float*)d_in[10];
  const float* qvec = (const float*)d_in[11];
  float* out = (float*)d_out;

  char* ws = (char*)d_ws;
  const size_t headElems = (size_t)B_ * H_ * S_ * DK_;   // 8M
  const size_t xElems    = (size_t)B_ * S_ * D_;         // 8M
  __bf16* qb = (__bf16*)ws; ws += headElems * sizeof(__bf16);
  __bf16* kb = (__bf16*)ws; ws += headElems * sizeof(__bf16);
  __bf16* vt = (__bf16*)ws; ws += headElems * sizeof(__bf16);  // [B,H,DK,S]
  float*  x  = (float*)ws;  ws += xElems * sizeof(float);
  __bf16* xb = (__bf16*)ws; ws += xElems * sizeof(__bf16);
  float*  a     = (float*)ws; ws += (size_t)B_ * S_ * sizeof(float);
  float*  alpha = (float*)ws; ws += (size_t)B_ * S_ * sizeof(float);

  qkv_proj_kernel<<<dim3(64, 16, 3), 256, 0, stream>>>(
      query, key_, value, Wq, bq, Wk, bk, Wv, bv, qb, kb, vt);

  attn_kernel<<<dim3(S_ / 64, H_, B_), 128, 0, stream>>>(qb, kb, vt, x, xb);

  pool_score_kernel<<<dim3(64), 256, 0, stream>>>(xb, Wa, ba, qvec, a);

  alpha_softmax_kernel<<<dim3(B_), 256, 0, stream>>>(a, alpha);

  pool_reduce_kernel<<<dim3(D_ / 256, B_), 256, 0, stream>>>(alpha, x, out);
}
